// PraxisRouter_24567212933862
// MI455X (gfx1250) — matrix operand order repaired and checked by reading, not compiled, not run on hardware
//
#include <hip/hip_runtime.h>
#include <hip/hip_bf16.h>

// ---------------------------------------------------------------------------
// Fused MoE router for MI455X (gfx1250, wave32, WMMA + async-LDS DMA).
//   GEMM1 (16384x2048 @ 2048x64) + gelu_new + GEMM2 (..x64 @ 64x64)
//   + gumbel-softmax + top-2 + expert histogram, fused in one pass over x.
// Memory-bound (~140MB @ 23.3TB/s ~= 6us); bf16 WMMA w/ f32 accum.
// w1 tiles double-buffered into LDS via GLOBAL_LOAD_ASYNC_TO_LDS_B128
// (ASYNCcnt), overlapping DMA with WMMA; one barrier per K-step.
// ---------------------------------------------------------------------------

typedef __attribute__((ext_vector_type(16))) __bf16 v16bf;
typedef __attribute__((ext_vector_type(8)))  float  v8f;

#define D_DIM  2048
#define E_DIM  64
#define BM     32      // tokens per block
#define KSTEP  32      // WMMA K depth (bf16)
#define EPSF   1e-10f

__device__ __forceinline__ float gelu_new_f(float v) {
    const float c = 0.7978845608028654f;  // sqrt(2/pi)
    float t = tanhf(c * (v + 0.044715f * v * v * v));
    return 0.5f * v * (1.0f + t);
}

// Async-copy 32 bytes (8 floats) global -> LDS, tracked by ASYNCcnt.
// Low 32 bits of a generic pointer to __shared__ are the LDS byte offset.
__device__ __forceinline__ void async_copy32(const float* gsrc, float* ldst) {
    unsigned loff = (unsigned)(uintptr_t)ldst;
    unsigned long long ga = (unsigned long long)(uintptr_t)gsrc;
    asm volatile("global_load_async_to_lds_b128 %0, %1, off"
                 :: "v"(loff), "v"(ga) : "memory");
    asm volatile("global_load_async_to_lds_b128 %0, %1, off offset:16"
                 :: "v"(loff), "v"(ga) : "memory");
}

__device__ __forceinline__ void wait_async0() {
    asm volatile("s_wait_asynccnt 0x0" ::: "memory");
}

extern "C" __global__ void __launch_bounds__(256)
init_counts_kernel(float* counts) {
    if (threadIdx.x < E_DIM) counts[threadIdx.x] = 0.0f;
}

extern "C" __global__ void __launch_bounds__(256)
router_fused_kernel(const float* __restrict__ x,  const float* __restrict__ w1,
                    const float* __restrict__ b1, const float* __restrict__ w2,
                    const float* __restrict__ b2, const float* __restrict__ u,
                    float* __restrict__ rp, float* __restrict__ idx_out,
                    float* __restrict__ counts) {
    __shared__ __bf16 lds_w2[E_DIM * E_DIM];        // [f][e] (transposed) 8KB
    __shared__ float  lds_w1f[2][KSTEP * E_DIM];    // [k][e] raw f32, 2x8KB
    __shared__ __bf16 lds_h [BM * E_DIM];           // [row][e]            4KB
    __shared__ float  lds_z [BM * E_DIM];           // [row][f]            8KB

    const int tid    = threadIdx.x;
    const int lane   = tid & 31;
    const int wave   = tid >> 5;
    const int mw     = wave >> 2;          // 0..1  (M tile)
    const int nw     = wave & 3;           // 0..3  (N tile)
    const int blockM = blockIdx.x * BM;
    const int l16    = lane & 15;
    const int hi     = lane >> 4;          // lane half

    // Stage w2 transposed into LDS as bf16: lds_w2[f*64+e] = w2[e][f]
    for (int i = tid; i < E_DIM * E_DIM; i += 256) {
        int e = i >> 6, f = i & 63;
        lds_w2[f * E_DIM + e] = (__bf16)w2[e * E_DIM + f];
    }

    // A-matrix 16-bit layout (ISA 7.12.2): lanes 0-15 -> K {0..7,16..23},
    // lanes 16-31 -> K {8..15,24..31}.  B dense 32x16 uses the same K order
    // as A.
    const int rowA = blockM + mw * 16 + l16;
    const int cA   = hi ? 8 : 0;
    const int kbB  = hi ? 16 : 0;
    const int colC = nw * 16 + l16;

    // ---------------- GEMM1: h = gelu(x @ w1 + b1) ----------------
    // Each thread DMAs 8 consecutive floats of the 32x64 w1 tile (contiguous
    // region of w1), so the async copy needs no transpose/conversion.
    const int stageOff = tid * 8;

    // prologue: DMA tile 0 into buffer 0
    async_copy32(w1 + stageOff, &lds_w1f[0][stageOff]);
    wait_async0();
    __syncthreads();

    v8f acc = {};
    const int NT = D_DIM / KSTEP;          // 64 K-steps
    for (int t = 0; t < NT; ++t) {
        const int cur = t & 1;
        const int k0  = t * KSTEP;

        // kick off DMA of next w1 tile into the other buffer
        if (t + 1 < NT)
            async_copy32(w1 + (size_t)(k0 + KSTEP) * E_DIM + stageOff,
                         &lds_w1f[1 - cur][stageOff]);

        // A fragment from global (fp32 -> bf16); prefetch next tile
        const float* xp = x + (size_t)rowA * D_DIM + k0 + cA;
        if (t + 1 < NT) __builtin_prefetch(xp + KSTEP, 0, 1);
        float4 a0 = *reinterpret_cast<const float4*>(xp);
        float4 a1 = *reinterpret_cast<const float4*>(xp + 4);
        float4 a2 = *reinterpret_cast<const float4*>(xp + 16);
        float4 a3 = *reinterpret_cast<const float4*>(xp + 20);
        v16bf af;
        af[0]  = (__bf16)a0.x; af[1]  = (__bf16)a0.y; af[2]  = (__bf16)a0.z; af[3]  = (__bf16)a0.w;
        af[4]  = (__bf16)a1.x; af[5]  = (__bf16)a1.y; af[6]  = (__bf16)a1.z; af[7]  = (__bf16)a1.w;
        af[8]  = (__bf16)a2.x; af[9]  = (__bf16)a2.y; af[10] = (__bf16)a2.z; af[11] = (__bf16)a2.w;
        af[12] = (__bf16)a3.x; af[13] = (__bf16)a3.y; af[14] = (__bf16)a3.z; af[15] = (__bf16)a3.w;

        // B fragment from LDS [k][e] f32 (K-strided, conflict-free across
        // lanes), convert to bf16
        const float* bp = &lds_w1f[cur][cA * E_DIM + colC];
        v16bf bf;
        #pragma unroll
        for (int j = 0; j < 8; ++j) { bf[j] = (__bf16)bp[j * E_DIM]; bf[8 + j] = (__bf16)bp[(16 + j) * E_DIM]; }

        acc = __builtin_amdgcn_wmma_f32_16x16x32_bf16(
            false, af, false, bf, (short)0, acc, false, false);

        // next tile's DMA must land + all waves done reading cur buffer
        wait_async0();
        __syncthreads();
    }

    // gelu_new + bias, stage h to LDS as bf16.
    // C/D layout: VGPR r -> M = r (lanes 0-15) or r+8 (lanes 16-31), N = l16.
    {
        const float bias = b1[colC];
        const int rbase = mw * 16 + hi * 8;
        #pragma unroll
        for (int r = 0; r < 8; ++r) {
            float v = acc[r] + bias;
            lds_h[(rbase + r) * E_DIM + colC] = (__bf16)gelu_new_f(v);
        }
    }
    __syncthreads();

    // ---------------- GEMM2: z = h @ w2 + b2 ----------------
    v8f acc2 = {};
    #pragma unroll
    for (int k0 = 0; k0 < E_DIM; k0 += KSTEP) {
        const __bf16* ap = &lds_h[(mw * 16 + l16) * E_DIM + k0 + cA];
        v16bf af;
        #pragma unroll
        for (int j = 0; j < 8; ++j) { af[j] = ap[j]; af[8 + j] = ap[16 + j]; }
        const __bf16* bp = &lds_w2[colC * E_DIM + k0 + cA];
        v16bf bf;
        #pragma unroll
        for (int j = 0; j < 8; ++j) { bf[j] = bp[j]; bf[8 + j] = bp[16 + j]; }
        acc2 = __builtin_amdgcn_wmma_f32_16x16x32_bf16(
            false, af, false, bf, (short)0, acc2, false, false);
    }
    {
        const float bias = b2[colC];
        const int rbase = mw * 16 + hi * 8;
        #pragma unroll
        for (int r = 0; r < 8; ++r)
            lds_z[(rbase + r) * E_DIM + colC] = acc2[r] + bias;
    }
    __syncthreads();

    // ---------------- gumbel-softmax + top-2 (one thread per token) --------
    if (tid < BM) {
        const int g = blockM + tid;
        const float* up = u + (size_t)g * E_DIM;
        float* zrow = &lds_z[tid * E_DIM];
        float m1 = -3.4e38f, m2 = -3.4e38f;
        int   i1 = 0, i2 = 0;
        for (int e = 0; e < E_DIM; ++e) {
            float uv  = up[e];
            float gum = -logf(-logf(uv + EPSF) + EPSF);
            float s   = zrow[e] + gum;          // tau = 1.0
            zrow[e] = s;
            if (s > m1)      { m2 = m1; i2 = i1; m1 = s; i1 = e; }
            else if (s > m2) { m2 = s;  i2 = e; }
        }
        float sum = 0.0f;
        for (int e = 0; e < E_DIM; ++e) sum += expf(zrow[e] - m1);
        float y1 = 1.0f / sum;                  // exp(0)/sum
        float y2 = expf(m2 - m1) / sum;
        float s1 = y1 + EPSF, s2 = y2 + EPSF;
        float inv = 1.0f / (s1 + s2);
        rp[g * 2 + 0]      = s1 * inv;
        rp[g * 2 + 1]      = s2 * inv;
        idx_out[g * 2 + 0] = (float)i1;
        idx_out[g * 2 + 1] = (float)i2;
        atomicAdd(&counts[i1], 1.0f);
        atomicAdd(&counts[i2], 1.0f);
    }
}

extern "C" __global__ void __launch_bounds__(64)
finalize_kernel(const float* __restrict__ counts, float* loss, float* temp) {
    if (threadIdx.x == 0) {
        float tot = 0.0f;
        for (int e = 0; e < E_DIM; ++e) tot += counts[e];
        const float target = 1.0f / (float)E_DIM;
        const float lt = logf(target);
        float a = 0.0f;
        for (int e = 0; e < E_DIM; ++e) {
            float p = counts[e] / tot;
            a += target * (lt - logf(p + EPSF));
        }
        *loss = a / (float)E_DIM;
        *temp = 1.0f;
    }
}

extern "C" void kernel_launch(void* const* d_in, const int* in_sizes, int n_in,
                              void* d_out, int out_size, void* d_ws, size_t ws_size,
                              hipStream_t stream) {
    const float* x  = (const float*)d_in[0];
    const float* w1 = (const float*)d_in[1];
    const float* b1 = (const float*)d_in[2];
    const float* w2 = (const float*)d_in[3];
    const float* b2 = (const float*)d_in[4];
    const float* u  = (const float*)d_in[5];

    const int tokens = in_sizes[0] / D_DIM;          // B*S = 16384

    // d_out layout (float), in reference return order:
    //   router_probs [tokens*2] | top_k_indices [tokens*2] | loss [1]
    //   | expert_counts [64] | temperature [1]
    float* out    = (float*)d_out;
    float* rp     = out;
    float* idxo   = out + (size_t)tokens * 2;
    float* loss   = out + (size_t)tokens * 4;
    float* counts = loss + 1;
    float* temp   = counts + E_DIM;

    init_counts_kernel<<<1, 256, 0, stream>>>(counts);
    router_fused_kernel<<<tokens / BM, 256, 0, stream>>>(
        x, w1, b1, w2, b2, u, rp, idxo, counts);
    finalize_kernel<<<1, 64, 0, stream>>>(counts, loss, temp);
}
